// FlairCNN_41790031790270
// MI455X (gfx1250) — compile-verified, matrix operand order checked by reading, not run on hardware
//
#include <hip/hip_runtime.h>
#include <hip/hip_bf16.h>

// ---------------------------------------------------------------------------
// BiLSTM (Flair embeddings) on MI455X / gfx1250.
//   K1: embedding gather -> f16 x [B,T,128]
//   K2: pack [Wih|Whh] -> f16 W [dir][2048][640]  (row-major [N,K], B-matrix ready)
//   K3: persistent WMMA LSTM: grid (4 batch tiles x 2 dirs), 16 waves/WG (512 thr).
//       Each wave owns 32 h-columns and all 4 gates for them (8 WMMA chains,
//       64 accum VGPRs); per step: 160x v_wmma_f32_16x16x32_f16, elementwise
//       gate math in registers, h exchanged via LDS. Mask pre-staged in LDS
//       so the serial time loop issues no global loads besides x_t/weights.
//   K4: boundary gather -> out [64,64,2048] f32
// ---------------------------------------------------------------------------

typedef __attribute__((ext_vector_type(8)))  _Float16 v8h;
typedef __attribute__((ext_vector_type(16))) _Float16 v16h;
typedef __attribute__((ext_vector_type(8)))  float    v8f;

#define B_   64
#define T_   512
#define E_   128
#define H_   512
#define G_   2048      // 4*H
#define KK_  640       // E + H
#define HP_  (H_ + 8)  // padded LDS row stride (halves), keeps 16B alignment
#define V_   300
#define W1_  65

__device__ __forceinline__ v16h frag16(const _Float16* p) {
    // lane supplies K chunks [k0..k0+7] and [k0+16..k0+23] -> 16 halves
    v8h lo = *(const v8h*)(p);
    v8h hi = *(const v8h*)(p + 16);
    return __builtin_shufflevector(lo, hi, 0,1,2,3,4,5,6,7,8,9,10,11,12,13,14,15);
}

__device__ __forceinline__ float sigmoidf_(float x) {
    return 1.0f / (1.0f + __expf(-x));
}

// ---------------- K1: embedding gather + f32->f16 ----------------
__global__ void embed_gather_kernel(const int* __restrict__ ids,
                                    const float* __restrict__ emb,
                                    _Float16* __restrict__ xh) {
    int i = blockIdx.x * blockDim.x + threadIdx.x;           // over B*T*E
    if (i >= B_ * T_ * E_) return;
    int bt = i >> 7;           // /E_
    int d  = i & (E_ - 1);
    int id = ids[bt];
    id = min(max(id, 0), V_ - 1);
    xh[i] = (_Float16)emb[id * E_ + d];
}

// ---------------- K2: pack weights [dir][G][KK] f16 ----------------
__global__ void weight_pack_kernel(const float* __restrict__ Wih_f,
                                   const float* __restrict__ Whh_f,
                                   const float* __restrict__ Wih_b,
                                   const float* __restrict__ Whh_b,
                                   _Float16* __restrict__ Wp) {
    int i = blockIdx.x * blockDim.x + threadIdx.x;           // over 2*G*KK
    if (i >= 2 * G_ * KK_) return;
    int dir = i / (G_ * KK_);
    int rem = i - dir * (G_ * KK_);
    int g = rem / KK_;
    int k = rem - g * KK_;
    const float* Wih = dir ? Wih_b : Wih_f;
    const float* Whh = dir ? Whh_b : Whh_f;
    float v = (k < E_) ? Wih[g * E_ + k] : Whh[g * H_ + (k - E_)];
    Wp[i] = (_Float16)v;
}

// ---------------- K3: persistent WMMA LSTM ----------------
// 512 threads = 16 waves; wave w owns h-columns [32w, 32w+32) as two 16-col
// subtiles (s = 0,1) for each of the 4 gates -> 8 accumulators per wave.
__global__ __launch_bounds__(512)
void lstm_wmma_kernel(const _Float16* __restrict__ xh,
                      const int* __restrict__ maskp,
                      const _Float16* __restrict__ Wp_all,
                      const float* __restrict__ bf,
                      const float* __restrict__ bb,
                      _Float16* __restrict__ hstream_all) {
    __shared__ __align__(16) _Float16 hprev[16 * HP_];   // 16.6 KB
    __shared__ float sm_mask[16 * T_];                    // 32 KB, [row][t]

    const int dir      = blockIdx.y;                 // 0 = fwd, 1 = bwd
    const int b0       = blockIdx.x * 16;            // batch tile base
    const bool reverse = (dir == 1);
    const _Float16* Wp   = Wp_all + (size_t)dir * G_ * KK_;
    const float*    bias = reverse ? bb : bf;
    _Float16* hstream    = hstream_all + (size_t)dir * B_ * T_ * H_;

    const int tid  = threadIdx.x;
    const int wave = tid >> 5;      // 0..15
    const int lane = tid & 31;
    const int l16  = lane & 15;
    const int hi   = lane >> 4;
    const int khi  = hi * 8;        // A/B fragment K sub-offset per ISA layout
    const int j0   = wave * 32;     // this wave's 32 h-columns
    const int rowR = hi << 3;       // C/D layout: row = v + 8*hi

    // zero initial hidden state
    for (int i = tid; i < 16 * HP_; i += 512) hprev[i] = (_Float16)0.0f;
    // pre-stage entire mask tile into LDS: removes all global mask traffic
    // (and its load latency) from the 512-iteration serial loop.
    for (int i = tid; i < 16 * T_; i += 512) {
        int r  = i >> 9;           // / T_
        int tt = i & (T_ - 1);
        sm_mask[i] = (float)maskp[(size_t)(b0 + r) * T_ + tt];
    }

    // per-lane bias for each (gate, subtile): column gt*512 + j0 + s*16 + l16
    float biasv[4][2];
    #pragma unroll
    for (int gt = 0; gt < 4; ++gt)
        #pragma unroll
        for (int s = 0; s < 2; ++s)
            biasv[gt][s] = bias[gt * 512 + j0 + s * 16 + l16];

    // single weight base pointer; per-tile row offset is a compile-time
    // immediate (gt*512 + s*16)*KK_ halves -> folds into the load offset.
    const _Float16* wbase = Wp + (size_t)(j0 + l16) * KK_ + khi;
    const _Float16* xrow  = xh + (size_t)(b0 + l16) * T_ * E_ + khi;
    const _Float16* hrow  = &hprev[l16 * HP_ + khi];

    v8f c[2], hreg[2];
    #pragma unroll
    for (int s = 0; s < 2; ++s)
        #pragma unroll
        for (int v = 0; v < 8; ++v) { c[s][v] = 0.0f; hreg[s][v] = 0.0f; }

    __syncthreads();

    for (int st = 0; st < T_; ++st) {
        const int t = reverse ? (T_ - 1 - st) : st;
        const _Float16* xp = xrow + (size_t)t * E_;

        v8f acc[4][2];
        // ---- K block 0 (x_t, K=0..31): C = inline 0, no accumulator init ----
        {
            v16h a = frag16(xp);
            #pragma unroll
            for (int gt = 0; gt < 4; ++gt)
                #pragma unroll
                for (int s = 0; s < 2; ++s) {
                    v8f z = {};
                    acc[gt][s] = __builtin_amdgcn_wmma_f32_16x16x32_f16(
                        false, a, false,
                        frag16(wbase + (gt * 512 + s * 16) * KK_),
                        (short)0, z, false, false);
                }
        }
        // ---- x_t contribution: K = 32..127 ----
        #pragma unroll 1
        for (int kb = 1; kb < 4; ++kb) {
            v16h a = frag16(xp + kb * 32);
            #pragma unroll
            for (int gt = 0; gt < 4; ++gt)
                #pragma unroll
                for (int s = 0; s < 2; ++s)
                    acc[gt][s] = __builtin_amdgcn_wmma_f32_16x16x32_f16(
                        false, a, false,
                        frag16(wbase + (gt * 512 + s * 16) * KK_ + kb * 32),
                        (short)0, acc[gt][s], false, false);
        }
        // ---- h_{t-1} contribution: K = 128..639 (A from LDS) ----
        #pragma unroll 1
        for (int kb = 0; kb < 16; ++kb) {
            v16h a = frag16(hrow + kb * 32);
            #pragma unroll
            for (int gt = 0; gt < 4; ++gt)
                #pragma unroll
                for (int s = 0; s < 2; ++s)
                    acc[gt][s] = __builtin_amdgcn_wmma_f32_16x16x32_f16(
                        false, a, false,
                        frag16(wbase + (gt * 512 + s * 16) * KK_ + E_ + kb * 32),
                        (short)0, acc[gt][s], false, false);
        }

        __syncthreads();   // all waves done reading hprev[t-1]

        // gate math; same (v,lane) slot across gate accumulators = same (row,col)
        #pragma unroll
        for (int s = 0; s < 2; ++s) {
            #pragma unroll
            for (int v = 0; v < 8; ++v) {
                float ig = sigmoidf_(acc[0][s][v] + biasv[0][s]);
                float fg = sigmoidf_(acc[1][s][v] + biasv[1][s]);
                float gg = tanhf(acc[2][s][v] + biasv[2][s]);
                float og = sigmoidf_(acc[3][s][v] + biasv[3][s]);
                int   r  = rowR + v;
                float m  = sm_mask[(r << 9) + t];
                float cn = fg * c[s][v] + ig * gg;
                float hn = og * tanhf(cn);
                bool  on = (m > 0.0f);
                float cu = on ? cn : c[s][v];
                float hu = on ? hn : hreg[s][v];
                c[s][v] = cu; hreg[s][v] = hu;
                int j = j0 + s * 16 + l16;
                hprev[r * HP_ + j] = (_Float16)hu;
                hstream[((size_t)(b0 + r) * T_ + t) * H_ + j] = (_Float16)(hu * m);
            }
        }
        __syncthreads();   // hprev[t] published before next step's reads
    }
}

// ---------------- K4: boundary gather ----------------
__global__ void gather_kernel(const int* __restrict__ boundary,
                              const _Float16* __restrict__ hstream_all,
                              float* __restrict__ out) {
    size_t idx = (size_t)blockIdx.x * blockDim.x + threadIdx.x;  // over 64*64*2048
    if (idx >= (size_t)B_ * 64 * 2048) return;
    int f = (int)(idx & 2047);          // feature within word: [hf_i|hb_i|hf_j|hb_j]
    int w = (int)((idx >> 11) & 63);
    int b = (int)(idx >> 17);
    int i_idx = boundary[b * W1_ + w];
    int j_idx = boundary[b * W1_ + w + 1];
    bool valid = (i_idx != -1) && (j_idx != -1);
    int tsel = (f < 1024) ? i_idx : j_idx;
    tsel = min(max(tsel, 0), T_ - 1);
    int dirsel = (f >> 9) & 1;
    int hcol = f & 511;
    float v = 0.0f;
    if (valid) {
        const _Float16* hs = hstream_all + (size_t)dirsel * B_ * T_ * H_;
        v = (float)hs[((size_t)b * T_ + tsel) * H_ + hcol];
    }
    out[idx] = v;
}

// ---------------- launch ----------------
extern "C" void kernel_launch(void* const* d_in, const int* in_sizes, int n_in,
                              void* d_out, int out_size, void* d_ws, size_t ws_size,
                              hipStream_t stream) {
    const int*   ids      = (const int*)d_in[0];
    const int*   mask     = (const int*)d_in[1];
    const int*   boundary = (const int*)d_in[2];
    const float* emb      = (const float*)d_in[3];
    const float* Wih_f    = (const float*)d_in[4];
    const float* Whh_f    = (const float*)d_in[5];
    const float* b_f      = (const float*)d_in[6];
    const float* Wih_b    = (const float*)d_in[7];
    const float* Whh_b    = (const float*)d_in[8];
    const float* b_b      = (const float*)d_in[9];
    float* out = (float*)d_out;

    // workspace layout (bytes):
    //   xh      : B*T*E*2           =  8 MiB   @ 0
    //   Wp      : 2*G*KK*2          ~  5.24 MiB @ 8 MiB
    //   hstream : 2*B*T*H*2         = 64 MiB   @ 16 MiB
    char* ws = (char*)d_ws;
    _Float16* xh      = (_Float16*)(ws);
    _Float16* Wp      = (_Float16*)(ws + (8ull  << 20));
    _Float16* hstream = (_Float16*)(ws + (16ull << 20));

    {   // K1
        int n = B_ * T_ * E_;
        embed_gather_kernel<<<(n + 255) / 256, 256, 0, stream>>>(ids, emb, xh);
    }
    {   // K2
        int n = 2 * G_ * KK_;
        weight_pack_kernel<<<(n + 255) / 256, 256, 0, stream>>>(Wih_f, Whh_f, Wih_b, Whh_b, Wp);
    }
    {   // K3: 4 batch tiles x 2 directions, 512 threads (16 waves)
        lstm_wmma_kernel<<<dim3(4, 2), 512, 0, stream>>>(xh, mask, Wp, b_f, b_b, hstream);
    }
    {   // K4
        size_t n = (size_t)B_ * 64 * 2048;
        gather_kernel<<<(unsigned)((n + 255) / 256), 256, 0, stream>>>(boundary, hstream, out);
    }
}
